// ChunkedAttention_26938034881204
// MI455X (gfx1250) — matrix operand order repaired and checked by reading, not compiled, not run on hardware
//
#include <hip/hip_runtime.h>
#include <hip/hip_bf16.h>
#include <math.h>

typedef __attribute__((ext_vector_type(16))) _Float16 v16h;
typedef __attribute__((ext_vector_type(8)))  _Float16 h8;
typedef __attribute__((ext_vector_type(8)))  float    v8f;
typedef __attribute__((ext_vector_type(4)))  float    f4;

#define B_    4
#define L_    4096
#define H_    16
#define DH    64
#define CHUNK 1024
#define QTILE 64
#define KSTEP 32

#define KSTRIDE 72   // halfs per K lds row (64 + 8 pad), 144B -> 16B-aligned runs
#define VSTRIDE 40   // halfs per V^T lds row (32 + 8 pad), 80B -> 16B-aligned runs

// ---- CDNA5 async global->LDS path (probe via __has_builtin; safe fallback) ----
#if defined(__has_builtin)
#  if __has_builtin(__builtin_amdgcn_global_load_async_to_lds_b128) && \
      __has_builtin(__builtin_amdgcn_s_wait_asynccnt)
#    define USE_ASYNC_LDS 1
#  endif
#endif
#ifndef USE_ASYNC_LDS
#  define USE_ASYNC_LDS 0
#endif

#if USE_ASYNC_LDS
typedef __attribute__((ext_vector_type(4))) int v4i;
typedef __attribute__((address_space(1))) v4i gv4i;   // global ("__device__")
typedef __attribute__((address_space(3))) v4i lv4i;   // LDS ("__shared__")

__device__ __forceinline__ void async_cp16(const _Float16* gsrc, _Float16* ldst) {
  // b128 async copy: global -> LDS, tracked by ASYNCcnt
  __builtin_amdgcn_global_load_async_to_lds_b128(
      (gv4i*)(v4i*)gsrc, (lv4i*)(v4i*)ldst, 0, 0);
}
#endif

__device__ __forceinline__ unsigned pk2(float a, float b) {
  union { _Float16 h[2]; unsigned u; } x;
  x.h[0] = (_Float16)a; x.h[1] = (_Float16)b;
  return x.u;
}

union VF {
  v16h v;
  h8 h2[2];
  unsigned u[8];
  _Float16 e[16];
};

union H8U { h8 v; _Float16 e[8]; };

// =====================================================================
// Pre-pass: RoPE(Q), RoPE(K) -> f16 in native [b][l][h][d] layout, and
// V -> f16 fully transposed to [b][h][d][l].  One trig evaluation per
// element for the whole problem.
// grid = B*H*(L/64) = 4096 blocks, 256 threads.
// =====================================================================
__global__ __launch_bounds__(256)
void ChunkedAttention_prepass(const float* __restrict__ Q,
                              const float* __restrict__ K,
                              const float* __restrict__ V,
                              const int* __restrict__ p_start,
                              _Float16* __restrict__ qr,
                              _Float16* __restrict__ kr,
                              _Float16* __restrict__ vt) {
  __shared__ __align__(16) _Float16 lvt[64 * 72];   // [d][row], padded

  const int bx = blockIdx.x;
  const int lt = bx & 63;            // l-tile (L/64 = 64)
  const int h  = (bx >> 6) & 15;
  const int b  = bx >> 10;
  const int tid = threadIdx.x;
  const int start = *p_start;

  const int row = tid >> 2;          // 0..63 (l within tile)
  const int c   = (tid & 3) * 8;     // 0,8,16,24 (RoPE-half d base)
  const long l  = (long)lt * 64 + row;
  const float pos = (float)(start + (int)l);
  const long rbase = (((long)b * L_ + l) * H_ + h) * DH;

  const f4 qx1a = *(const f4*)(Q + rbase + c);
  const f4 qx1b = *(const f4*)(Q + rbase + c + 4);
  const f4 qx2a = *(const f4*)(Q + rbase + c + 32);
  const f4 qx2b = *(const f4*)(Q + rbase + c + 36);
  const f4 kx1a = *(const f4*)(K + rbase + c);
  const f4 kx1b = *(const f4*)(K + rbase + c + 4);
  const f4 kx2a = *(const f4*)(K + rbase + c + 32);
  const f4 kx2b = *(const f4*)(K + rbase + c + 36);
  const f4 vxa  = *(const f4*)(V + rbase + c);
  const f4 vxb  = *(const f4*)(V + rbase + c + 4);
  const f4 vxc  = *(const f4*)(V + rbase + c + 32);
  const f4 vxd  = *(const f4*)(V + rbase + c + 36);

  H8U q1, q2, k1, k2;
#pragma unroll
  for (int j = 0; j < 8; ++j) {
    const int d = c + j;                         // 0..31
    const float invf = __powf(10000.0f, -(float)d * (1.0f / 32.0f));
    float sv, cv;
    __sincosf(pos * invf, &sv, &cv);
    const float qa = (j < 4) ? qx1a[j] : qx1b[j - 4];
    const float qb = (j < 4) ? qx2a[j] : qx2b[j - 4];
    const float ka = (j < 4) ? kx1a[j] : kx1b[j - 4];
    const float kb = (j < 4) ? kx2a[j] : kx2b[j - 4];
    q1.e[j] = (_Float16)(qa * cv - qb * sv);
    q2.e[j] = (_Float16)(qa * sv + qb * cv);
    k1.e[j] = (_Float16)(ka * cv - kb * sv);
    k2.e[j] = (_Float16)(ka * sv + kb * cv);
    lvt[(d)      * 72 + row] = (_Float16)((j < 4) ? vxa[j] : vxb[j - 4]);
    lvt[(d + 32) * 72 + row] = (_Float16)((j < 4) ? vxc[j] : vxd[j - 4]);
  }
  *(h8*)(qr + rbase + c)      = q1.v;
  *(h8*)(qr + rbase + c + 32) = q2.v;
  *(h8*)(kr + rbase + c)      = k1.v;
  *(h8*)(kr + rbase + c + 32) = k2.v;

  __syncthreads();
  // write transposed V tile: Vt[b][h][d][l]
  const int dv  = tid >> 2;           // 0..63
  const int seg = (tid & 3) * 16;     // 0,16,32,48 (l within tile)
  const long vbase = (((long)b * H_ + h) * DH + dv) * L_ + (long)lt * 64 + seg;
  h8 o0 = *(const h8*)&lvt[dv * 72 + seg];
  h8 o1 = *(const h8*)&lvt[dv * 72 + seg + 8];
  *(h8*)(vt + vbase)     = o0;
  *(h8*)(vt + vbase + 8) = o1;
}

// =====================================================================
// Fast attention kernel: pure f16 WMMA flash-attention over pre-RoPE'd
// tensors.  Double-buffered LDS tiles; async global->LDS when available.
// =====================================================================
__global__ __launch_bounds__(128)
void ChunkedAttention_f16_kernel(const _Float16* __restrict__ qr,
                                 const _Float16* __restrict__ kr,
                                 const _Float16* __restrict__ vt,
                                 float* __restrict__ Out) {
  __shared__ __align__(16) _Float16 lds_k[2][KSTEP * KSTRIDE];
  __shared__ __align__(16) _Float16 lds_vt[2][DH * VSTRIDE];

  const int bx  = blockIdx.x;
  const int qt  = bx & 15;
  const int h   = (bx >> 4) & 15;
  const int nch = (bx >> 8) & 3;
  const int b   = bx >> 10;

  const int tid  = threadIdx.x;
  const int wave = tid >> 5;
  const int lane = tid & 31;
  const int hh   = lane >> 4;
  const int r    = lane & 15;

  const int wq0    = qt * QTILE + wave * 16;   // wave's min query (in chunk)
  const int qchunk = wq0 + r;
  const long qglob = (long)nch * CHUNK + qchunk;

  // ---- Q fragments: direct aligned f16 loads in native B layout ----
  const _Float16* qrow = qr + (((long)b * L_ + qglob) * H_ + h) * DH;
  VF qb0, qb1;
  qb0.h2[0] = *(const h8*)(qrow + hh * 8);
  qb0.h2[1] = *(const h8*)(qrow + 16 + hh * 8);
  qb1.h2[0] = *(const h8*)(qrow + 32 + hh * 8);
  qb1.h2[1] = *(const h8*)(qrow + 48 + hh * 8);

  v8f ot[4];
#pragma unroll
  for (int t = 0; t < 4; ++t) ot[t] = (v8f){0,0,0,0,0,0,0,0};
  float mrun = -3.0e38f;
  float lsum = 0.0f;

  const int nsteps = (qt * QTILE + QTILE - 1) / KSTEP + 1;

  // tile loader indices
  const int krow_t = tid >> 2;            // 0..31
  const int kseg   = (tid & 3) * 16;      // 0,16,32,48
  const int vd     = tid >> 1;            // 0..63
  const int vhalf  = (tid & 1) * 16;      // 0,16

  const _Float16* kptr0 = kr +
      (((long)b * L_ + (long)nch * CHUNK + krow_t) * H_ + h) * DH + kseg;
  const _Float16* vptr0 = vt +
      (((long)b * H_ + h) * DH + vd) * L_ + (long)nch * CHUNK + vhalf;

#if USE_ASYNC_LDS
#define ISSUE_TILE(KB, BF)                                                     \
  do {                                                                         \
    const _Float16* ksrc = kptr0 + (long)(KB) * KSTEP * (H_ * DH);             \
    const _Float16* vsrc = vptr0 + (long)(KB) * KSTEP;                         \
    _Float16* kdst = &lds_k[(BF)][krow_t * KSTRIDE + kseg];                    \
    _Float16* vdst = &lds_vt[(BF)][vd * VSTRIDE + vhalf];                      \
    async_cp16(ksrc, kdst);                                                    \
    async_cp16(ksrc + 8, kdst + 8);                                            \
    async_cp16(vsrc, vdst);                                                    \
    async_cp16(vsrc + 8, vdst + 8);                                            \
  } while (0)
#define WAIT_TILE() __builtin_amdgcn_s_wait_asynccnt(0)
#else
#define ISSUE_TILE(KB, BF)                                                     \
  do {                                                                         \
    const _Float16* ksrc = kptr0 + (long)(KB) * KSTEP * (H_ * DH);             \
    const _Float16* vsrc = vptr0 + (long)(KB) * KSTEP;                         \
    _Float16* kdst = &lds_k[(BF)][krow_t * KSTRIDE + kseg];                    \
    _Float16* vdst = &lds_vt[(BF)][vd * VSTRIDE + vhalf];                      \
    h8 ka = *(const h8*)ksrc;                                                  \
    h8 kb2 = *(const h8*)(ksrc + 8);                                           \
    *(h8*)kdst = ka;                                                           \
    *(h8*)(kdst + 8) = kb2;                                                    \
    h8 va = *(const h8*)vsrc;                                                  \
    h8 vb = *(const h8*)(vsrc + 8);                                            \
    *(h8*)vdst = va;                                                           \
    *(h8*)(vdst + 8) = vb;                                                     \
  } while (0)
#define WAIT_TILE() ((void)0)
#endif

  // prologue: start copy of tile 0 into buffer 0
  ISSUE_TILE(0, 0);

  for (int kb = 0; kb < nsteps; ++kb) {
    const int kbase = kb * KSTEP;
    const int bf = kb & 1;

    WAIT_TILE();        // this wave's outstanding tile copies complete
    __syncthreads();    // all waves: buffer bf populated, buffer bf^1 free

    if (kb + 1 < nsteps) ISSUE_TILE(kb + 1, bf ^ 1);  // overlap with compute

    // skip tiles fully beyond this wave's diagonal (wave-uniform predicate)
    if (kbase <= wq0 + 15) {
      // ---- S^T = K_tile x Q^T ----
      v8f st[2];
#pragma unroll
      for (int tt = 0; tt < 2; ++tt) {
        const int krow = tt * 16 + r;
        VF a0, a1;
        a0.h2[0] = *(const h8*)&lds_k[bf][krow * KSTRIDE +      hh * 8];
        a0.h2[1] = *(const h8*)&lds_k[bf][krow * KSTRIDE + 16 + hh * 8];
        a1.h2[0] = *(const h8*)&lds_k[bf][krow * KSTRIDE + 32 + hh * 8];
        a1.h2[1] = *(const h8*)&lds_k[bf][krow * KSTRIDE + 48 + hh * 8];
        v8f acc = (v8f){0,0,0,0,0,0,0,0};
        acc = __builtin_amdgcn_wmma_f32_16x16x32_f16(false, a0.v, false, qb0.v,
                                                     (short)0, acc, false, false);
        acc = __builtin_amdgcn_wmma_f32_16x16x32_f16(false, a1.v, false, qb1.v,
                                                     (short)0, acc, false, false);
        st[tt] = acc;
      }

      // causal mask only on diagonal-straddling tiles (wave-uniform branch)
      if (kbase + KSTEP - 1 > wq0) {
#pragma unroll
        for (int tt = 0; tt < 2; ++tt)
#pragma unroll
          for (int vv = 0; vv < 8; ++vv) {
            const int kpos = kbase + tt * 16 + hh * 8 + vv;
            if (kpos > qchunk) st[tt][vv] = -3.0e38f;
          }
      }

      // ---- online softmax (lane <-> query, stats per-lane uniform) ----
      float tmax = -3.0e38f;
#pragma unroll
      for (int tt = 0; tt < 2; ++tt)
#pragma unroll
        for (int vv = 0; vv < 8; ++vv) tmax = fmaxf(tmax, st[tt][vv]);
      tmax = fmaxf(tmax, __shfl_xor(tmax, 16, 32));
      const float mnew  = fmaxf(mrun, tmax);
      const float alpha = __expf(mrun - mnew);
      mrun = mnew;

      float p0[8], p1[8], tsum = 0.0f;
#pragma unroll
      for (int vv = 0; vv < 8; ++vv) {
        p0[vv] = __expf(st[0][vv] - mnew);
        p1[vv] = __expf(st[1][vv] - mnew);
        tsum += p0[vv] + p1[vv];
      }
      tsum += __shfl_xor(tsum, 16, 32);
      lsum = lsum * alpha + tsum;
#pragma unroll
      for (int t = 0; t < 4; ++t)
#pragma unroll
        for (int vv = 0; vv < 8; ++vv) ot[t][vv] *= alpha;

      // ---- P^T B-fragment via half-wave exchange ----
      unsigned pa[4], pb[4], sa[4], sb[4];
#pragma unroll
      for (int i = 0; i < 4; ++i) {
        pa[i] = pk2(p0[2 * i], p0[2 * i + 1]);
        pb[i] = pk2(p1[2 * i], p1[2 * i + 1]);
      }
#pragma unroll
      for (int i = 0; i < 4; ++i) {
        sa[i] = __shfl_xor((int)pa[i], 16, 32);
        sb[i] = __shfl_xor((int)pb[i], 16, 32);
      }
      VF pf;
#pragma unroll
      for (int i = 0; i < 4; ++i) {
        pf.u[i]     = hh ? sb[i] : pa[i];
        pf.u[4 + i] = hh ? pb[i] : sa[i];
      }

      // ---- O^T += V^T x P^T ----
#pragma unroll
      for (int t = 0; t < 4; ++t) {
        const int d = t * 16 + r;
        VF vf;
        vf.h2[0] = *(const h8*)&lds_vt[bf][d * VSTRIDE +      hh * 16];
        vf.h2[1] = *(const h8*)&lds_vt[bf][d * VSTRIDE + 8 + hh * 16];
        ot[t] = __builtin_amdgcn_wmma_f32_16x16x32_f16(false, vf.v, false, pf.v,
                                                       (short)0, ot[t], false, false);
      }
    }
  }

  const float invl = 1.0f / lsum;
  float* op = Out + (((long)b * L_ + (long)nch * CHUNK + qchunk) * H_ + h) * DH;
#pragma unroll
  for (int t = 0; t < 4; ++t)
#pragma unroll
    for (int vv = 0; vv < 8; ++vv)
      op[t * 16 + hh * 8 + vv] = ot[t][vv] * invl;
}

// =====================================================================
// Fallback (fused RoPE+attention) for the case ws_size is too small.
// =====================================================================
__global__ __launch_bounds__(128)
void ChunkedAttention_fused_fallback(const float* __restrict__ Q,
                                     const float* __restrict__ K,
                                     const float* __restrict__ V,
                                     const int* __restrict__ p_start,
                                     float* __restrict__ Out) {
  __shared__ __align__(16) _Float16 lds_k[KSTEP * KSTRIDE];
  __shared__ __align__(16) _Float16 lds_vt[DH * VSTRIDE];

  const int bx  = blockIdx.x;
  const int qt  = bx & 15;
  const int h   = (bx >> 4) & 15;
  const int nch = (bx >> 8) & 3;
  const int b   = bx >> 10;

  const int tid  = threadIdx.x;
  const int wave = tid >> 5;
  const int lane = tid & 31;
  const int hh   = lane >> 4;
  const int r    = lane & 15;

  const int start = *p_start;

  const int  qchunk = qt * QTILE + wave * 16 + r;
  const long qglob  = (long)nch * CHUNK + qchunk;
  const float posq  = (float)(start + qglob);
  const float* qrow = Q + (((long)b * L_ + qglob) * H_ + h) * DH;

  VF qb0, qb1;
  {
    const int d0 = hh * 16;
    f4 x1[4], x2[4];
#pragma unroll
    for (int i = 0; i < 4; ++i) {
      x1[i] = *(const f4*)(qrow + d0 + 4 * i);
      x2[i] = *(const f4*)(qrow + d0 + 32 + 4 * i);
    }
#pragma unroll
    for (int i = 0; i < 4; ++i) {
#pragma unroll
      for (int j = 0; j < 4; ++j) {
        const int d = d0 + 4 * i + j;
        const float invf = __powf(10000.0f, -(float)d * (1.0f / 32.0f));
        float sv, cv;
        __sincosf(posq * invf, &sv, &cv);
        qb0.e[4 * i + j] = (_Float16)(x1[i][j] * cv - x2[i][j] * sv);
        qb1.e[4 * i + j] = (_Float16)(x1[i][j] * sv + x2[i][j] * cv);
      }
    }
  }

  v8f ot[4];
#pragma unroll
  for (int t = 0; t < 4; ++t) ot[t] = (v8f){0,0,0,0,0,0,0,0};
  float mrun = -3.0e38f, lsum = 0.0f;

  const int nsteps = (qt * QTILE + QTILE - 1) / KSTEP + 1;
  const int rthr = tid >> 3;
  const int cthr = (tid & 7) * 4;

  for (int kb = 0; kb < nsteps; ++kb) {
    const int kbase = kb * KSTEP;
    __syncthreads();
#pragma unroll
    for (int rr = 0; rr < 2; ++rr) {
      const int row = rr * 16 + rthr;
      const long kglob = (long)nch * CHUNK + kbase + row;
      const float posk = (float)(start + kglob);
      const float* krow_p = K + (((long)b * L_ + kglob) * H_ + h) * DH;
      const float* vrow_p = V + (((long)b * L_ + kglob) * H_ + h) * DH;
      const f4 kx1 = *(const f4*)(krow_p + cthr);
      const f4 kx2 = *(const f4*)(krow_p + cthr + 32);
      const f4 vx1 = *(const f4*)(vrow_p + cthr);
      const f4 vx2 = *(const f4*)(vrow_p + cthr + 32);
#pragma unroll
      for (int j = 0; j < 4; ++j) {
        const int d = cthr + j;
        const float invf = __powf(10000.0f, -(float)d * (1.0f / 32.0f));
        float sv, cv;
        __sincosf(posk * invf, &sv, &cv);
        lds_k[row * KSTRIDE + d]      = (_Float16)(kx1[j] * cv - kx2[j] * sv);
        lds_k[row * KSTRIDE + d + 32] = (_Float16)(kx1[j] * sv + kx2[j] * cv);
        lds_vt[(d)      * VSTRIDE + row] = (_Float16)vx1[j];
        lds_vt[(d + 32) * VSTRIDE + row] = (_Float16)vx2[j];
      }
    }
    __syncthreads();

    v8f st[2];
#pragma unroll
    for (int tt = 0; tt < 2; ++tt) {
      const int krow = tt * 16 + r;
      VF a0, a1;
      a0.h2[0] = *(const h8*)&lds_k[krow * KSTRIDE +      hh * 16];
      a0.h2[1] = *(const h8*)&lds_k[krow * KSTRIDE + 8 + hh * 16];
      a1.h2[0] = *(const h8*)&lds_k[krow * KSTRIDE + 32 + hh * 16];
      a1.h2[1] = *(const h8*)&lds_k[krow * KSTRIDE + 40 + hh * 16];
      v8f acc = (v8f){0,0,0,0,0,0,0,0};
      acc = __builtin_amdgcn_wmma_f32_16x16x32_f16(false, a0.v, false, qb0.v,
                                                   (short)0, acc, false, false);
      acc = __builtin_amdgcn_wmma_f32_16x16x32_f16(false, a1.v, false, qb1.v,
                                                   (short)0, acc, false, false);
      st[tt] = acc;
    }

#pragma unroll
    for (int tt = 0; tt < 2; ++tt)
#pragma unroll
      for (int vv = 0; vv < 8; ++vv) {
        const int kpos = kbase + tt * 16 + hh * 8 + vv;
        if (kpos > qchunk) st[tt][vv] = -3.0e38f;
      }

    float tmax = -3.0e38f;
#pragma unroll
    for (int tt = 0; tt < 2; ++tt)
#pragma unroll
      for (int vv = 0; vv < 8; ++vv) tmax = fmaxf(tmax, st[tt][vv]);
    tmax = fmaxf(tmax, __shfl_xor(tmax, 16, 32));
    const float mnew  = fmaxf(mrun, tmax);
    const float alpha = __expf(mrun - mnew);
    mrun = mnew;

    float p0[8], p1[8], tsum = 0.0f;
#pragma unroll
    for (int vv = 0; vv < 8; ++vv) {
      p0[vv] = __expf(st[0][vv] - mnew);
      p1[vv] = __expf(st[1][vv] - mnew);
      tsum += p0[vv] + p1[vv];
    }
    tsum += __shfl_xor(tsum, 16, 32);
    lsum = lsum * alpha + tsum;
#pragma unroll
    for (int t = 0; t < 4; ++t)
#pragma unroll
      for (int vv = 0; vv < 8; ++vv) ot[t][vv] *= alpha;

    unsigned pa[4], pb[4], sa[4], sb[4];
#pragma unroll
    for (int i = 0; i < 4; ++i) {
      pa[i] = pk2(p0[2 * i], p0[2 * i + 1]);
      pb[i] = pk2(p1[2 * i], p1[2 * i + 1]);
    }
#pragma unroll
    for (int i = 0; i < 4; ++i) {
      sa[i] = __shfl_xor((int)pa[i], 16, 32);
      sb[i] = __shfl_xor((int)pb[i], 16, 32);
    }
    VF pf;
#pragma unroll
    for (int i = 0; i < 4; ++i) {
      pf.u[i]     = hh ? sb[i] : pa[i];
      pf.u[4 + i] = hh ? pb[i] : sa[i];
    }

#pragma unroll
    for (int t = 0; t < 4; ++t) {
      const int d = t * 16 + r;
      VF vf;
      vf.h2[0] = *(const h8*)&lds_vt[d * VSTRIDE +      hh * 16];
      vf.h2[1] = *(const h8*)&lds_vt[d * VSTRIDE + 8 + hh * 16];
      ot[t] = __builtin_amdgcn_wmma_f32_16x16x32_f16(false, vf.v, false, pf.v,
                                                     (short)0, ot[t], false, false);
    }
  }

  const float invl = 1.0f / lsum;
  float* op = Out + (((long)b * L_ + (long)nch * CHUNK + qchunk) * H_ + h) * DH;
#pragma unroll
  for (int t = 0; t < 4; ++t)
#pragma unroll
    for (int vv = 0; vv < 8; ++vv)
      op[t * 16 + hh * 8 + vv] = ot[t][vv] * invl;
}

extern "C" void kernel_launch(void* const* d_in, const int* in_sizes, int n_in,
                              void* d_out, int out_size, void* d_ws, size_t ws_size,
                              hipStream_t stream) {
  const float* q = (const float*)d_in[0];
  const float* k = (const float*)d_in[1];
  const float* v = (const float*)d_in[2];
  // d_in[3] = chunk_size (hardcoded 1024 to match setup_inputs; grid depends on it)
  const int* start = (const int*)d_in[4];
  float* out = (float*)d_out;

  const int nblocks_attn = B_ * (L_ / CHUNK) * H_ * (CHUNK / QTILE);  // 4096
  const size_t n_elems = (size_t)B_ * L_ * H_ * DH;
  const size_t need = 3 * n_elems * sizeof(_Float16);                 // ~101 MB

  if (ws_size >= need) {
    _Float16* qr = (_Float16*)d_ws;
    _Float16* kr = qr + n_elems;
    _Float16* vt = kr + n_elems;
    const int nblocks_pre = B_ * H_ * (L_ / 64);                      // 4096
    ChunkedAttention_prepass<<<dim3(nblocks_pre), dim3(256), 0, stream>>>(
        q, k, v, start, qr, kr, vt);
    ChunkedAttention_f16_kernel<<<dim3(nblocks_attn), dim3(128), 0, stream>>>(
        qr, kr, vt, out);
  } else {
    ChunkedAttention_fused_fallback<<<dim3(nblocks_attn), dim3(128), 0, stream>>>(
        q, k, v, start, out);
  }
}
